// FP8Linear_64458869178673
// MI455X (gfx1250) — 8-bit matrix operand order repaired and checked by reading, compiled, not run on hardware
//
#include <hip/hip_runtime.h>

typedef __attribute__((ext_vector_type(16))) int   v16i;
typedef __attribute__((ext_vector_type(8)))  float v8f;

#define GM 8192
#define GN 2048
#define GK 2048
#define KT (GK / 128)

#define XQ_OFF 256
#define XQ_BYTES ((size_t)GM * GK)
#define WQ_OFF (XQ_OFF + XQ_BYTES)
#define WQ_BYTES ((size_t)GN * GK)

__device__ __forceinline__ unsigned f32_to_e4m3_sw(float f) {
    unsigned u    = __float_as_uint(f);
    unsigned sign = (u >> 24) & 0x80u;
    unsigned a    = u & 0x7FFFFFFFu;
    if (a > 0x43E00000u) a = 0x43E00000u;
    if (a < 0x3C800000u) {
        float t = __uint_as_float(a) * 512.0f;
        unsigned m = (unsigned)(int)rintf(t);
        return sign | m;
    }
    unsigned mant = a & 0x7FFFFFu;
    unsigned exp  = a >> 23;
    unsigned m3   = mant >> 20;
    unsigned rem  = mant & 0xFFFFFu;
    m3 += (rem > 0x80000u) || ((rem == 0x80000u) && (m3 & 1));
    exp += (m3 >> 3);
    m3  &= 7u;
    unsigned code = ((exp - 120u) << 3) | m3;
    if (code > 0x7Eu) code = 0x7Eu;
    return sign | code;
}

__device__ __forceinline__ unsigned pack4_e4m3(float a, float b, float c, float d) {
#if __has_builtin(__builtin_amdgcn_cvt_pk_fp8_f32)
    int v = 0;
    v = __builtin_amdgcn_cvt_pk_fp8_f32(a, b, v, false);
    v = __builtin_amdgcn_cvt_pk_fp8_f32(c, d, v, true);
    return (unsigned)v;
#else
    return f32_to_e4m3_sw(a) | (f32_to_e4m3_sw(b) << 8) |
           (f32_to_e4m3_sw(c) << 16) | (f32_to_e4m3_sw(d) << 24);
#endif
}

__global__ void init_kernel(unsigned* ws_u) { ws_u[0] = 0u; ws_u[1] = 0u; }

__global__ void amax_kernel(const float* __restrict__ p, size_t n4, unsigned* dst) {
    size_t i      = (size_t)blockIdx.x * blockDim.x + threadIdx.x;
    size_t stride = (size_t)gridDim.x * blockDim.x;
    float m = 0.0f;
    const float4* p4 = (const float4*)p;
    for (; i < n4; i += stride) {
        float4 v = p4[i];
        m = fmaxf(m, fmaxf(fmaxf(fabsf(v.x), fabsf(v.y)),
                           fmaxf(fabsf(v.z), fabsf(v.w))));
    }
    #pragma unroll
    for (int off = 16; off > 0; off >>= 1)
        m = fmaxf(m, __shfl_xor(m, off, 32));
    if ((threadIdx.x & 31) == 0)
        atomicMax(dst, __float_as_uint(m));
}

__global__ void scales_kernel(const unsigned* ws_u, float* ws_f) {
    float ax = __uint_as_float(ws_u[0]);
    float aw = __uint_as_float(ws_u[1]);
    const float inv = 1.0f / (448.0f * 0.8f);
    float xs = ax * inv;
    float wscale = fmaxf(ax, aw) * inv;
    ws_f[2] = xs;
    ws_f[3] = wscale;
    ws_f[4] = xs * wscale;
    ws_f[5] = 1.0f / xs;
    ws_f[6] = 1.0f / wscale;
}

__global__ void quant_x_kernel(const float* __restrict__ x, const float* __restrict__ ws_f,
                               unsigned char* __restrict__ xq) {
    const float rcp = ws_f[5];
    size_t c = (size_t)blockIdx.x * blockDim.x + threadIdx.x;
    size_t nchunks = XQ_BYTES / 8;
    if (c >= nchunks) return;
    unsigned fb   = (unsigned)(c >> 8);
    unsigned w    = (unsigned)(c & 255u);
    unsigned lane = w >> 3;
    unsigned j    = w & 7u;
    unsigned mt = fb / KT, kt = fb % KT;
    unsigned m  = mt * 16 + (lane & 15u);
    unsigned k  = kt * 128 + 32u * (j >> 1) + 8u * (j & 1u) + 16u * (lane >> 4);
    const float4* src = (const float4*)(x + (size_t)m * GK + k);
    float4 v0 = src[0], v1 = src[1];
    unsigned lo = pack4_e4m3(v0.x * rcp, v0.y * rcp, v0.z * rcp, v0.w * rcp);
    unsigned hi = pack4_e4m3(v1.x * rcp, v1.y * rcp, v1.z * rcp, v1.w * rcp);
    ((uint2*)xq)[c] = make_uint2(lo, hi);
}

__global__ void quant_w_kernel(const float* __restrict__ wgt, const float* __restrict__ ws_f,
                               unsigned char* __restrict__ wq) {
    const float rcp = ws_f[6];
    size_t c = (size_t)blockIdx.x * blockDim.x + threadIdx.x;
    size_t nchunks = WQ_BYTES / 16;
    if (c >= nchunks) return;
    unsigned fb   = (unsigned)(c >> 7);
    unsigned w    = (unsigned)(c & 127u);
    unsigned lane = w >> 2;
    unsigned q    = w & 3u;
    unsigned nt = fb / KT, kt = fb % KT;
    unsigned n  = nt * 16 + (lane & 15u);
    unsigned k  = kt * 128 + 32u * q + 16u * (lane >> 4);
    const float4* src = (const float4*)(wgt + (size_t)n * GK + k);
    unsigned d[4];
    #pragma unroll
    for (int t = 0; t < 4; ++t) {
        float4 v = src[t];
        d[t] = pack4_e4m3(v.x * rcp, v.y * rcp, v.z * rcp, v.w * rcp);
    }
    ((uint4*)wq)[c] = make_uint4(d[0], d[1], d[2], d[3]);
}

__global__ __launch_bounds__(256)
void gemm_fp8_kernel(const unsigned char* __restrict__ Aq,
                     const unsigned char* __restrict__ Bq,
                     const float* __restrict__ ws_f,
                     const float* __restrict__ bias,
                     float* __restrict__ out) {
    const int lane = threadIdx.x & 31;
    const int wave = threadIdx.x >> 5;
    const int wm = wave >> 1;
    const int wn = wave & 1;
    const int mt0 = blockIdx.y * 16 + wm * 4;
    const int nt0 = blockIdx.x * 8  + wn * 4;

    const unsigned char* Abase = Aq + lane * 64;
    const unsigned char* Bbase = Bq + lane * 64;
    #define AFRAG(mt, kt) (*(const v16i*)(Abase + (((size_t)(mt) * KT + (kt)) << 11)))
    #define BFRAG(nt, kt) (*(const v16i*)(Bbase + (((size_t)(nt) * KT + (kt)) << 11)))

    v8f acc[4][4] = {};
    v16i b0, b1, b2, b3, a0, a1;

    b0 = BFRAG(nt0 + 0, 0);
    b1 = BFRAG(nt0 + 1, 0);
    b2 = BFRAG(nt0 + 2, 0);
    b3 = BFRAG(nt0 + 3, 0);
    a0 = AFRAG(mt0 + 0, 0);

    for (int kt = 0; kt < KT; ++kt) {
        a1 = AFRAG(mt0 + 1, kt);
        acc[0][0] = __builtin_amdgcn_wmma_f32_16x16x128_fp8_fp8(a0, b0, (short)0, acc[0][0], false, false);
        acc[0][1] = __builtin_amdgcn_wmma_f32_16x16x128_fp8_fp8(a0, b1, (short)0, acc[0][1], false, false);
        acc[0][2] = __builtin_amdgcn_wmma_f32_16x16x128_fp8_fp8(a0, b2, (short)0, acc[0][2], false, false);
        acc[0][3] = __builtin_amdgcn_wmma_f32_16x16x128_fp8_fp8(a0, b3, (short)0, acc[0][3], false, false);

        a0 = AFRAG(mt0 + 2, kt);
        acc[1][0] = __builtin_amdgcn_wmma_f32_16x16x128_fp8_fp8(a1, b0, (short)0, acc[1][0], false, false);
        acc[1][1] = __builtin_amdgcn_wmma_f32_16x16x128_fp8_fp8(a1, b1, (short)0, acc[1][1], false, false);
        acc[1][2] = __builtin_amdgcn_wmma_f32_16x16x128_fp8_fp8(a1, b2, (short)0, acc[1][2], false, false);
        acc[1][3] = __builtin_amdgcn_wmma_f32_16x16x128_fp8_fp8(a1, b3, (short)0, acc[1][3], false, false);

        a1 = AFRAG(mt0 + 3, kt);
        acc[2][0] = __builtin_amdgcn_wmma_f32_16x16x128_fp8_fp8(a0, b0, (short)0, acc[2][0], false, false);
        acc[2][1] = __builtin_amdgcn_wmma_f32_16x16x128_fp8_fp8(a0, b1, (short)0, acc[2][1], false, false);
        acc[2][2] = __builtin_amdgcn_wmma_f32_16x16x128_fp8_fp8(a0, b2, (short)0, acc[2][2], false, false);
        acc[2][3] = __builtin_amdgcn_wmma_f32_16x16x128_fp8_fp8(a0, b3, (short)0, acc[2][3], false, false);

        const int ktn = (kt + 1 < KT) ? kt + 1 : kt;
        a0 = AFRAG(mt0 + 0, ktn);
        acc[3][0] = __builtin_amdgcn_wmma_f32_16x16x128_fp8_fp8(a1, b0, (short)0, acc[3][0], false, false);
        acc[3][1] = __builtin_amdgcn_wmma_f32_16x16x128_fp8_fp8(a1, b1, (short)0, acc[3][1], false, false);
        b0 = BFRAG(nt0 + 0, ktn);
        b1 = BFRAG(nt0 + 1, ktn);
        acc[3][2] = __builtin_amdgcn_wmma_f32_16x16x128_fp8_fp8(a1, b2, (short)0, acc[3][2], false, false);
        acc[3][3] = __builtin_amdgcn_wmma_f32_16x16x128_fp8_fp8(a1, b3, (short)0, acc[3][3], false, false);
        b2 = BFRAG(nt0 + 2, ktn);
        b3 = BFRAG(nt0 + 3, ktn);
    }
    #undef AFRAG
    #undef BFRAG

    const float s   = ws_f[4];
    const int  col  = lane & 15;
    const int  roff = (lane >> 4) * 8;
    #pragma unroll
    for (int i = 0; i < 4; ++i) {
        const int mbase = (mt0 + i) * 16 + roff;
        #pragma unroll
        for (int j = 0; j < 4; ++j) {
            const int n = (nt0 + j) * 16 + col;
            const float bz = bias[n];
            #pragma unroll
            for (int v = 0; v < 8; ++v)
                out[(size_t)(mbase + v) * GN + n] = acc[i][j][v] * s + bz;
        }
    }
}

extern "C" void kernel_launch(void* const* d_in, const int* in_sizes, int n_in,
                              void* d_out, int out_size, void* d_ws, size_t ws_size,
                              hipStream_t stream) {
    const float* x    = (const float*)d_in[0];
    const float* wgt  = (const float*)d_in[1];
    const float* bias = (const float*)d_in[2];
    float* out = (float*)d_out;

    unsigned*      ws_u = (unsigned*)d_ws;
    float*         ws_f = (float*)d_ws;
    unsigned char* xq   = (unsigned char*)d_ws + XQ_OFF;
    unsigned char* wq   = (unsigned char*)d_ws + WQ_OFF;

    init_kernel<<<1, 1, 0, stream>>>(ws_u);

    amax_kernel<<<2048, 256, 0, stream>>>(x,   (size_t)GM * GK / 4, ws_u + 0);
    amax_kernel<<<512,  256, 0, stream>>>(wgt, (size_t)GN * GK / 4, ws_u + 1);

    scales_kernel<<<1, 1, 0, stream>>>(ws_u, ws_f);

    {
        size_t nchunks = XQ_BYTES / 8;
        quant_x_kernel<<<(unsigned)((nchunks + 255) / 256), 256, 0, stream>>>(x, ws_f, xq);
    }
    {
        size_t nchunks = WQ_BYTES / 16;
        quant_w_kernel<<<(unsigned)((nchunks + 255) / 256), 256, 0, stream>>>(wgt, ws_f, wq);
    }

    dim3 grid(GN / 128, GM / 256);
    gemm_fp8_kernel<<<grid, 256, 0, stream>>>(xq, wq, ws_f, bias, out);
}
